// DilatedToothSegNet_69621419868375
// MI455X (gfx1250) — matrix operand order repaired and checked by reading, not compiled, not run on hardware
//
#include <hip/hip_runtime.h>
#include <hip/hip_bf16.h>

// ---------------- problem constants ----------------
#define B_    2
#define N_    8192
#define K_    32        // neighbors
#define NCLS  17
#define NP_   (B_ * N_)          // 16384 points
#define E_    (NP_ * K_)         // 524288 edges
#define NEG_  0.2f
#define EPS_  1e-5f

typedef __attribute__((ext_vector_type(16))) _Float16 v16h;
typedef __attribute__((ext_vector_type(8)))  _Float16 v8h;
typedef __attribute__((ext_vector_type(8)))  float    v8f;

// ---------------- utility kernels ----------------
__global__ void k_zero(float* __restrict__ p, int n) {
  int i = blockIdx.x * blockDim.x + threadIdx.x;
  if (i < n) p[i] = 0.f;
}

// x: (B,27,N) -> c (NP,12), n (NP,12), a (NP,3)  (f32, for kNN + gather)
__global__ void k_split(const float* __restrict__ x, float* __restrict__ c,
                        float* __restrict__ nr, float* __restrict__ ax) {
  int t = blockIdx.x * blockDim.x + threadIdx.x;
  if (t >= NP_) return;
  int b = t / N_, n = t % N_;
  const float* xb = x + (size_t)b * 27 * N_;
#pragma unroll
  for (int ch = 0; ch < 12; ++ch) c[(size_t)t * 12 + ch]  = xb[(size_t)ch * N_ + n];
#pragma unroll
  for (int ch = 0; ch < 12; ++ch) nr[(size_t)t * 12 + ch] = xb[(size_t)(12 + ch) * N_ + n];
#pragma unroll
  for (int ch = 0; ch < 3;  ++ch) ax[(size_t)t * 3 + ch]  = xb[(size_t)(24 + ch) * N_ + n];
}

// ---------------- kNN: per-thread query, LDS candidate tiles ----------------
template<int C>
__global__ __launch_bounds__(256) void k_knn(const float* __restrict__ pts,
                                             int* __restrict__ idxOut) {
  constexpr int NK = K_ + 1;              // keep self, drop later
  __shared__ float tile[64 * C];
  int b = blockIdx.y;
  int q = blockIdx.x * 256 + threadIdx.x;
  const float* pb = pts + (size_t)b * N_ * C;

  float qf[C];
#pragma unroll
  for (int c = 0; c < C; ++c) qf[c] = pb[(size_t)q * C + c];

  float dl[NK];
  int   il[NK];
#pragma unroll
  for (int t = 0; t < NK; ++t) { dl[t] = 3.0e38f; il[t] = 0; }

  for (int t0 = 0; t0 < N_; t0 += 64) {
    __syncthreads();
    for (int i = threadIdx.x; i < 64 * C; i += 256)
      tile[i] = pb[(size_t)t0 * C + i];
    __syncthreads();
    for (int ci = 0; ci < 64; ++ci) {
      float d = 0.f;
#pragma unroll
      for (int c = 0; c < C; ++c) {
        float df = qf[c] - tile[ci * C + c];
        d += df * df;
      }
      int cand = t0 + ci;
      if (d < dl[NK - 1]) {
        bool done = false;
#pragma unroll
        for (int j = NK - 1; j >= 1; --j) {
          float prev = dl[j - 1];
          if (d < prev)      { dl[j] = prev; il[j] = il[j - 1]; }
          else if (!done)    { dl[j] = d; il[j] = cand; done = true; }
        }
        if (!done) { dl[0] = d; il[0] = cand; }
      }
    }
  }
  int* op = idxOut + ((size_t)b * N_ + q) * K_;
#pragma unroll
  for (int t = 1; t < NK; ++t) op[t - 1] = il[t];
}

// ---------------- edge-feature gather -> padded f16 ----------------
// MODE 1: 'global'        out[e][k] = k<C0 ? x[b,j,k] : 0
// MODE 2: 'local_global'  out[e][k] = k<C0 ? x[j]-x[n] : (k<2C0 ? x[n][k-C0] : 0)
template<int MODE>
__global__ void k_gather_h(const float* __restrict__ pts, const int* __restrict__ nbr,
                           _Float16* __restrict__ out, int C0, int KP) {
  int t = blockIdx.x * blockDim.x + threadIdx.x;
  if (t >= E_ * KP) return;          // E_*KP <= 50M, fits int
  int e = t / KP, k = t % KP;
  int b  = e / (N_ * K_);
  int pn = (e / K_) % N_;
  int j  = nbr[e];
  const float* xb = pts + (size_t)b * N_ * C0;
  float v = 0.f;
  if (MODE == 1) {
    if (k < C0) v = xb[(size_t)j * C0 + k];
  } else {
    if (k < C0)          v = xb[(size_t)j * C0 + k] - xb[(size_t)pn * C0 + k];
    else if (k < 2 * C0) v = xb[(size_t)pn * C0 + (k - C0)];
  }
  out[t] = (_Float16)v;
}

// ---------------- weight staging: f32 (Nout x Kc) -> f16 (rowsPad x KCPAD), zero padded
__global__ void k_wconv(const float* __restrict__ W, _Float16* __restrict__ Wh,
                        int rowsPad, int KCPAD, int Nout, int Kc) {
  int t = blockIdx.x * blockDim.x + threadIdx.x;
  if (t >= rowsPad * KCPAD) return;
  int n = t / KCPAD, k = t % KCPAD;
  float v = (n < Nout && k < Kc) ? W[(size_t)n * Kc + k] : 0.f;
  Wh[t] = (_Float16)v;
}

// ---------------- WMMA GEMM on pre-staged padded f16 operands ----------------
// A: M x KCPAD (f16, zero padded).  W: rowsPad x KCPAD, rowsPad >= NoutPad.
// C: M x ldC, all NoutPad columns written (pad cols get exact zeros from zero weights).
template<int KCPAD, bool F16S>
__global__ __launch_bounds__(256) void k_gemm(
    const _Float16* __restrict__ Ah, const _Float16* __restrict__ Wh,
    void* __restrict__ Cp, float* __restrict__ ssum, float* __restrict__ ssq,
    int M, int NoutPad, int Nout, int ldC) {
  int lane = threadIdx.x & 31;
  int wid  = (blockIdx.x * blockDim.x + threadIdx.x) >> 5;
  int ntN  = NoutPad >> 4;
  if (wid >= (M >> 4) * ntN) return;       // wave-uniform exit, EXEC all-ones for WMMA
  int tm = wid / ntN, tn = wid % ntN;

  int am = tm * 16 + (lane & 15);          // A row of this lane
  int cn = tn * 16 + (lane & 15);          // output column of this lane
  int kb8  = (lane >> 4) * 8;              // A-frag K base (ISA 16-bit A layout)
  int kb16 = (lane >> 4) * 16;             // unused; B frag uses kb8 like A

  size_t aRow = (size_t)am * KCPAD;
  size_t bRow = (size_t)cn * KCPAD;

  v8f acc = {};
#pragma unroll
  for (int k0 = 0; k0 < KCPAD; k0 += 32) {
    if (KCPAD > 64) __builtin_prefetch(Ah + aRow + k0 + 128, 0, 1);
    const v8h* ap = (const v8h*)(Ah + aRow + k0 + kb8);
    const v8h* bp = (const v8h*)(Wh + bRow + k0 + kb8);
    v8h a0 = ap[0];        // k = kb8 .. kb8+7
    v8h a1 = ap[2];        // k = kb8+16 .. kb8+23
    v8h b0 = bp[0];        // k = kb8 .. kb8+7
    v8h b1 = bp[2];        // k = kb8+16 .. kb8+23
    v16h af = __builtin_shufflevector(a0, a1, 0,1,2,3,4,5,6,7,8,9,10,11,12,13,14,15);
    v16h bf = __builtin_shufflevector(b0, b1, 0,1,2,3,4,5,6,7,8,9,10,11,12,13,14,15);
    acc = __builtin_amdgcn_wmma_f32_16x16x32_f16(false, af, false, bf,
                                                 (short)0, acc, false, false);
  }

  float ls = 0.f, lq = 0.f;
#pragma unroll
  for (int g = 0; g < 8; ++g) {            // C/D layout: M = g + 8*(lane>=16), N = lane&15
    int rm = tm * 16 + g + 8 * (lane >> 4);
    float v = acc[g];
    if (F16S) ((_Float16*)Cp)[(size_t)rm * ldC + cn] = (_Float16)v;
    else      ((float*)Cp)[(size_t)rm * ldC + cn] = v;
    ls += v; lq += v * v;
  }
  if (ssum != nullptr && cn < Nout) {
    atomicAdd(&ssum[cn], ls);
    atomicAdd(&ssq[cn],  lq);
  }
}

// ---------------- batchnorm helpers ----------------
__global__ void k_bn_finalize(const float* __restrict__ sum, const float* __restrict__ sq,
                              const float* __restrict__ gamma, const float* __restrict__ beta,
                              float* __restrict__ scale, float* __restrict__ bias,
                              int C, float invCnt) {
  int c = blockIdx.x * blockDim.x + threadIdx.x;
  if (c >= C) return;
  float m = sum[c] * invCnt;
  float v = sq[c] * invCnt - m * m;
  float s = gamma[c] * rsqrtf(v + EPS_);
  scale[c] = s;
  bias[c]  = beta[c] - m * s;
}

// in-place BN + lrelu on padded f16 activations (pad columns stay zero)
__global__ void k_bn_lrelu_h(_Float16* __restrict__ x, const float* __restrict__ sc,
                             const float* __restrict__ bi, int n, int C, int CP) {
  int i = blockIdx.x * blockDim.x + threadIdx.x;
  if (i >= n) return;
  int c = i % CP;
  if (c >= C) return;
  float v = (float)x[i] * sc[c] + bi[c];
  x[i] = (_Float16)(v > 0.f ? v : NEG_ * v);
}

// h: (NP_, K_, 48) f16 -> bn+lrelu -> max over K_ -> o: (NP_, 48) f32
__global__ void k_bn_lrelu_max_h(const _Float16* __restrict__ h, const float* __restrict__ sc,
                                 const float* __restrict__ bi, float* __restrict__ o) {
  int t = blockIdx.x * blockDim.x + threadIdx.x;
  if (t >= NP_ * 48) return;
  int p = t / 48, c = t % 48;
  float s = sc[c], b = bi[c];
  float mx = -3.4e38f;
  for (int kk = 0; kk < K_; ++kk) {
    float v = (float)h[((size_t)p * K_ + kk) * 48 + c] * s + b;
    v = v > 0.f ? v : NEG_ * v;
    mx = fmaxf(mx, v);
  }
  o[t] = mx;
}

// concat three f32 (NP x 48) -> padded f16 (NP x CP)
__global__ void k_cat3_f32h(const float* __restrict__ a, const float* __restrict__ b,
                            const float* __restrict__ c, _Float16* __restrict__ o,
                            int ce, int CP) {
  int t = blockIdx.x * blockDim.x + threadIdx.x;
  if (t >= NP_ * CP) return;
  int r = t / CP, ch = t % CP;
  float v = 0.f;
  if (ch < ce)           v = a[(size_t)r * ce + ch];
  else if (ch < 2 * ce)  v = b[(size_t)r * ce + (ch - ce)];
  else if (ch < 3 * ce)  v = c[(size_t)r * ce + (ch - 2 * ce)];
  o[t] = (_Float16)v;
}

// concat three f16 (NP x ce) -> f16 (NP x 3*ce)
__global__ void k_cat3_hh(const _Float16* __restrict__ a, const _Float16* __restrict__ b,
                          const _Float16* __restrict__ c, _Float16* __restrict__ o, int ce) {
  int tot = 3 * ce;
  int t = blockIdx.x * blockDim.x + threadIdx.x;
  if (t >= NP_ * tot) return;
  int r = t / tot, ch = t % tot;
  _Float16 v;
  if (ch < ce)           v = a[(size_t)r * ce + ch];
  else if (ch < 2 * ce)  v = b[(size_t)r * ce + (ch - ce)];
  else                   v = c[(size_t)r * ce + (ch - 2 * ce)];
  o[t] = v;
}

// raw: (B*N, 32) f32 (only 17 valid) -> out: (B, 17, N)
__global__ void k_out_tr(const float* __restrict__ raw, float* __restrict__ out) {
  int t = blockIdx.x * blockDim.x + threadIdx.x;
  if (t >= B_ * NCLS * N_) return;
  int b = t / (NCLS * N_);
  int r = t % (NCLS * N_);
  int o = r / N_, n = r % N_;
  out[t] = raw[((size_t)b * N_ + n) * 32 + o];
}

// ---------------- host orchestration ----------------
static inline int divUp(long a, long b) { return (int)((a + b - 1) / b); }
static inline int pad32(int v) { return (v + 31) & ~31; }

// JAX tree-flatten (alphabetical, recursive) indices for the inputs dict:
enum {
  IN_FPS = 0,
  PA1 = 1, PA2 = 7, PA3 = 13, PC1 = 19, PC2 = 25, PC3 = 31,
  PFA = 37,
  PH1P1 = 40, PH1P2 = 43, PH2P1 = 46, PH2P2 = 49,
  PLOCA = 52, PLOCC = 55, PLOCN = 58,
  PLOCAL = 61,
  PN1 = 64, PN2 = 70, PN3 = 76,
  POUTW = 82, IN_X = 83
};

extern "C" void kernel_launch(void* const* d_in, const int* in_sizes, int n_in,
                              void* d_out, int out_size, void* d_ws, size_t ws_size,
                              hipStream_t stream) {
  (void)in_sizes; (void)n_in; (void)out_size; (void)ws_size;
  auto P = [&](int i) { return (const float*)d_in[i]; };

  // workspace carve-up (float units; every block a multiple of 8 floats -> 16B aligned)
  float* ws = (float*)d_ws;
  size_t off = 0;
  auto F32 = [&](size_t n) { float* p = ws + off; off += n; return p; };
  auto F16 = [&](size_t nh) { return (_Float16*)F32(nh / 2); };  // nh always even

  float* xc = F32((size_t)NP_ * 12);
  float* xn = F32((size_t)NP_ * 12);
  float* xa = F32((size_t)NP_ * 3 + 8);
  int* idx1 = (int*)F32(E_);
  int* idx2 = (int*)F32(E_);
  int* idx3 = (int*)F32(E_);
  float* c1 = F32((size_t)NP_ * 48); float* c2 = F32((size_t)NP_ * 48); float* c3 = F32((size_t)NP_ * 48);
  float* n1 = F32((size_t)NP_ * 48); float* n2 = F32((size_t)NP_ * 48); float* n3 = F32((size_t)NP_ * 48);
  float* a1 = F32((size_t)NP_ * 48); float* a2 = F32((size_t)NP_ * 48); float* a3 = F32((size_t)NP_ * 48);
  _Float16* eh   = F16((size_t)E_ * 96);   // gathered edge features
  _Float16* h1h  = F16((size_t)E_ * 64);   // first-layer hidden
  _Float16* h2h  = F16((size_t)E_ * 48);   // second-layer hidden
  _Float16* cat160  = F16((size_t)NP_ * 160);
  _Float16* lch  = F16((size_t)NP_ * 256);
  _Float16* lnh  = F16((size_t)NP_ * 256);
  _Float16* lah  = F16((size_t)NP_ * 256);
  _Float16* cat768h = F16((size_t)NP_ * 768);
  _Float16* fhA = F16((size_t)NP_ * 512);
  _Float16* fhB = F16((size_t)NP_ * 512);
  _Float16* Wh  = F16((size_t)512 * 768);  // staged weights (max: local 512x768)
  float* outraw = F32((size_t)NP_ * 32);
  float* ssum = F32(1024); float* ssq = F32(1024);
  float* sscale = F32(1024); float* sbias = F32(1024);

  auto zero = [&](float* p, int n) {
    k_zero<<<divUp(n, 256), 256, 0, stream>>>(p, n);
  };
  auto wconv = [&](const float* W, int rowsPad, int KCPAD, int Nout, int Kc) {
    k_wconv<<<divUp((long)rowsPad * KCPAD, 256), 256, 0, stream>>>(W, Wh, rowsPad, KCPAD, Nout, Kc);
  };
  auto gemm = [&](int KCPAD, bool f16s, const _Float16* Ah, void* Cp,
                  float* sum_, float* sq_, int M, int NoutPad, int Nout, int ldC) {
    long waves = (long)(M / 16) * (NoutPad / 16);
    int blocks = divUp(waves, 8);
#define GEMM_CASE(KP) case KP: \
    if (f16s) k_gemm<KP, true ><<<blocks, 256, 0, stream>>>(Ah, Wh, Cp, sum_, sq_, M, NoutPad, Nout, ldC); \
    else      k_gemm<KP, false><<<blocks, 256, 0, stream>>>(Ah, Wh, Cp, sum_, sq_, M, NoutPad, Nout, ldC); \
    break;
    switch (KCPAD) {
      GEMM_CASE(32) GEMM_CASE(64) GEMM_CASE(96) GEMM_CASE(160)
      GEMM_CASE(256) GEMM_CASE(512) GEMM_CASE(768)
      default: break;
    }
#undef GEMM_CASE
  };
  auto bnfin = [&](const float* g, const float* b, int C, float cnt) {
    k_bn_finalize<<<divUp(C, 256), 256, 0, stream>>>(ssum, ssq, g, b, sscale, sbias, C, 1.f / cnt);
  };
  auto bnact_h = [&](_Float16* x, long n, int C, int CP) {
    k_bn_lrelu_h<<<divUp(n, 256), 256, 0, stream>>>(x, sscale, sbias, (int)n, C, CP);
  };

  // graph_conv: feat f32 (NP_,C0) + idx -> out f32 (NP_,48)
  auto graph_conv = [&](const float* feat, int C0, int mode, const int* idx,
                        int base, int hid, float* outF) {
    const float* W1 = P(base), *W2 = P(base + 1);
    const float* b1 = P(base + 2), *b2 = P(base + 3);
    const float* g1 = P(base + 4), *g2 = P(base + 5);
    int Kc1 = (mode == 2) ? 2 * C0 : C0;
    int KP1 = pad32(Kc1);
    int hidP = pad32(hid);
    // gather edges into padded f16
    if (mode == 2)
      k_gather_h<2><<<divUp((long)E_ * KP1, 256), 256, 0, stream>>>(feat, idx, eh, C0, KP1);
    else
      k_gather_h<1><<<divUp((long)E_ * KP1, 256), 256, 0, stream>>>(feat, idx, eh, C0, KP1);
    // layer 1
    wconv(W1, hidP, KP1, hid, Kc1);
    zero(ssum, hid); zero(ssq, hid);
    gemm(KP1, true, eh, h1h, ssum, ssq, E_, hidP, hid, hidP);
    bnfin(g1, b1, hid, (float)E_);
    bnact_h(h1h, (long)E_ * hidP, hid, hidP);
    // layer 2
    wconv(W2, 48, hidP, 48, hid);
    zero(ssum, 48); zero(ssq, 48);
    gemm(hidP, true, h1h, h2h, ssum, ssq, E_, 48, 48, 48);
    bnfin(g2, b2, 48, (float)E_);
    k_bn_lrelu_max_h<<<divUp((long)NP_ * 48, 256), 256, 0, stream>>>(h2h, sscale, sbias, outF);
  };

  // pointwise: inH f16 (NP_,KCPADin) -> outH f16 (NP_,cout), BN+lrelu fused after
  auto pw = [&](const _Float16* inH, int cin, int KCPADin, int cout, int base,
                _Float16* outH) {
    const float* W = P(base), *bb = P(base + 1), *gg = P(base + 2);
    wconv(W, cout, KCPADin, cout, cin);            // cout is a multiple of 32 here
    zero(ssum, cout); zero(ssq, cout);
    gemm(KCPADin, true, inH, outH, ssum, ssq, NP_, cout, cout, cout);
    bnfin(gg, bb, cout, (float)NP_);
    bnact_h(outH, (long)NP_ * cout, cout, cout);
  };

  const float* x = P(IN_X);
  k_split<<<divUp(NP_, 256), 256, 0, stream>>>(x, xc, xn, xa);

  // stage 1
  k_knn<12><<<dim3(N_ / 256, B_), 256, 0, stream>>>(xc, idx1);
  graph_conv(xc, 12, 2, idx1, PC1, 32, c1);
  graph_conv(xn, 12, 1, idx1, PN1, 32, n1);
  graph_conv(xa, 3,  1, idx1, PA1, 32, a1);
  // stage 2
  k_knn<48><<<dim3(N_ / 256, B_), 256, 0, stream>>>(c1, idx2);
  graph_conv(c1, 48, 2, idx2, PC2, 48, c2);
  graph_conv(n1, 48, 1, idx2, PN2, 48, n2);
  graph_conv(a1, 48, 1, idx2, PA2, 48, a2);
  // stage 3
  k_knn<48><<<dim3(N_ / 256, B_), 256, 0, stream>>>(c2, idx3);
  graph_conv(c2, 48, 2, idx3, PC3, 48, c3);
  graph_conv(n2, 48, 1, idx3, PN3, 48, n3);
  graph_conv(a2, 48, 1, idx3, PA3, 48, a3);

  // per-branch local embeddings: concat(48,48,48) -> f16 padded 160 -> pw 256
  k_cat3_f32h<<<divUp((long)NP_ * 160, 256), 256, 0, stream>>>(c1, c2, c3, cat160, 48, 160);
  pw(cat160, 144, 160, 256, PLOCC, lch);
  k_cat3_f32h<<<divUp((long)NP_ * 160, 256), 256, 0, stream>>>(n1, n2, n3, cat160, 48, 160);
  pw(cat160, 144, 160, 256, PLOCN, lnh);
  k_cat3_f32h<<<divUp((long)NP_ * 160, 256), 256, 0, stream>>>(a1, a2, a3, cat160, 48, 160);
  pw(cat160, 144, 160, 256, PLOCA, lah);

  // fuse
  k_cat3_hh<<<divUp((long)NP_ * 768, 256), 256, 0, stream>>>(lch, lnh, lah, cat768h, 256);
  pw(cat768h, 768, 768, 512, PLOCAL, fhA);
  pw(fhA, 512, 512, 512, PFA,  fhB);
  pw(fhB, 512, 512, 256, PH1P1, fhA);
  pw(fhA, 256, 256, 256, PH1P2, fhB);
  pw(fhB, 256, 256, 256, PH2P1, fhA);
  pw(fhA, 256, 256, 256, PH2P2, fhB);

  // classifier head (no BN/activation), f32 store, ld=32, then transpose
  wconv(P(POUTW), 32, 256, NCLS, 256);
  gemm(256, false, fhB, outraw, nullptr, nullptr, NP_, 32, NCLS, 32);
  k_out_tr<<<divUp((long)B_ * NCLS * N_, 256), 256, 0, stream>>>(outraw, (float*)d_out);
}
